// MOE_90520730730735
// MI455X (gfx1250) — compile-verified, matrix operand order checked by reading, not run on hardware
//
#include <hip/hip_runtime.h>
#include <hip/hip_bf16.h>
#include <stdint.h>

// ---------------- problem constants (match reference setup_inputs) ----------
constexpr int B_   = 8;
constexpr int T_   = 2048;
constexpr int C_   = 1024;
constexpr int H_   = 2048;
constexpr int E_   = 8;
constexpr int TOPK = 2;
constexpr int CAP  = 640;
constexpr int RPE  = B_ * CAP;      // 5120 rows per expert
constexpr int LDSTR = 48;           // LDS row stride (elems), 96B (16B aligned)
constexpr int TILE_E = 64 * LDSTR;  // elems per 64x32 LDS tile (padded)

typedef __attribute__((ext_vector_type(16))) __bf16        v16bf;
typedef __attribute__((ext_vector_type(8)))  float         v8f;
typedef __attribute__((ext_vector_type(4)))  unsigned int  v4u;
typedef __attribute__((ext_vector_type(8)))  int           v8i_;
typedef __attribute__((ext_vector_type(4)))  int           v4i_;

#if __has_builtin(__builtin_amdgcn_tensor_load_to_lds) && \
    __has_builtin(__builtin_amdgcn_s_wait_tensorcnt)
#define USE_TDM 1
#endif

union Frag { v16bf v; uint4 q[2]; };

__device__ __forceinline__ unsigned short f2bf(float f) {
  unsigned int u = __float_as_uint(f);
  unsigned int r = u + 0x7FFFu + ((u >> 16) & 1u);   // round-to-nearest-even
  return (unsigned short)(r >> 16);
}

// Load one 16x32 bf16 WMMA fragment from LDS laid out row-major with the
// documented K-pair interleave: VGPR0..3 hold K = {0..7}+8*half,
// VGPR4..7 hold K = {16..23}+8*half  ->  two aligned ds_load_b128.
__device__ __forceinline__ v16bf load_frag(const unsigned short* __restrict__ p) {
  Frag f;
  f.q[0] = *(const uint4*)(p);
  f.q[1] = *(const uint4*)(p + 16);
  return f.v;
}

#ifdef USE_TDM
// Issue one TDM load of a 64-row x 32-elem bf16 tile (row stride = stride_elems)
// into LDS at lds_off, padding LDS rows to 96B (8 dwords pad per 16 dwords) so
// the landed layout matches LDSTR=48.  D# packing per CDNA5 ISA.
// clang-23 lane: 6-arg builtin (uint32x4, int32x8, int32x4, int32x4, int32x8, i32).
__device__ __forceinline__ void tdm_load_tile(unsigned long long gaddr,
                                              unsigned lds_off,
                                              unsigned stride_elems) {
  v4u g0;
  g0.x = 1u;                                           // count=1, user mode
  g0.y = lds_off;                                      // lds_addr
  g0.z = (unsigned)(gaddr & 0xFFFFFFFFull);            // global_addr lo
  g0.w = (unsigned)((gaddr >> 32) & 0x1FFFFFFull)      // global_addr hi (57b)
       | (2u << 30);                                   // type = 2 ("image")
  v8i_ g1;
  g1[0] = (int)((1u << 16)        // data_size = 1 -> 2 bytes
              | (1u << 20)        // pad_enable
              | (3u << 22)        // pad_interval: every 16 dwords (64B row)
              | (7u << 25));      // pad_amount: 8 dwords (32B) -> 96B row
  g1[1] = (int)((stride_elems & 0xFFFFu) << 16);            // tensor_dim0 lo16
  g1[2] = (int)(((stride_elems >> 16) & 0xFFFFu)            // tensor_dim0 hi16
              | (0x4000u << 16));                           // tensor_dim1 lo16
  g1[3] = (int)(32u << 16);                                 // tile_dim0 = 32
  g1[4] = 64;                                               // tile_dim1 = 64
  g1[5] = (int)stride_elems;                                // dim0_stride lo32
  g1[6] = 0;
  g1[7] = 0;
  v4i_ z4 = {0, 0, 0, 0};
  v8i_ z8 = {0, 0, 0, 0, 0, 0, 0, 0};
  __builtin_amdgcn_tensor_load_to_lds(g0, g1, z4, z4, z8, 0);
}
#endif

// ---------------------------------------------------------------------------
// 1) f32 [K][N] -> bf16 [N][K] transposed conversion (done once per weight).
// ---------------------------------------------------------------------------
__global__ __launch_bounds__(256) void moe_cvt_bf16_T(
    const float* __restrict__ src, unsigned short* __restrict__ dst,
    int K, int N) {
  __shared__ unsigned short tile[32][33];
  int e = blockIdx.z;
  const float* s = src + (size_t)e * K * N;
  unsigned short* d = dst + (size_t)e * K * N;
  int k0 = blockIdx.y * 32, n0 = blockIdx.x * 32;
#pragma unroll
  for (int i = 0; i < 4; ++i) {
    int id = threadIdx.x + i * 256;
    int r = id >> 5, c = id & 31;                 // r: k, c: n (coalesced read)
    tile[c][r] = f2bf(s[(size_t)(k0 + r) * N + n0 + c]);
  }
  __syncthreads();
#pragma unroll
  for (int i = 0; i < 4; ++i) {
    int id = threadIdx.x + i * 256;
    int r = id >> 5, c = id & 31;                 // r: n, c: k (coalesced write)
    d[(size_t)(n0 + r) * K + k0 + c] = tile[r][c];
  }
}

// ---------------------------------------------------------------------------
// 2) Router: logits -> softmax -> top-2.  One wave32 per token.
// ---------------------------------------------------------------------------
__global__ __launch_bounds__(256) void moe_router(
    const float* __restrict__ x, const float* __restrict__ wr,
    float* __restrict__ topk_p, int* __restrict__ topk_i) {
  int wave = blockIdx.x * 8 + (threadIdx.x >> 5);
  int lane = threadIdx.x & 31;
  int b = wave / T_, t = wave % T_;
  const float* xr = x + (size_t)(b * T_ + t) * C_;

  float acc[E_];
#pragma unroll
  for (int e = 0; e < E_; ++e) acc[e] = 0.f;

  for (int i = 0; i < C_ / 32; ++i) {
    int c = lane + i * 32;
    float xv = xr[c];
    const float4* w4 = (const float4*)(wr + (size_t)c * E_);
    float4 w0 = w4[0], w1 = w4[1];
    acc[0] = fmaf(xv, w0.x, acc[0]); acc[1] = fmaf(xv, w0.y, acc[1]);
    acc[2] = fmaf(xv, w0.z, acc[2]); acc[3] = fmaf(xv, w0.w, acc[3]);
    acc[4] = fmaf(xv, w1.x, acc[4]); acc[5] = fmaf(xv, w1.y, acc[5]);
    acc[6] = fmaf(xv, w1.z, acc[6]); acc[7] = fmaf(xv, w1.w, acc[7]);
  }
#pragma unroll
  for (int e = 0; e < E_; ++e)
#pragma unroll
    for (int off = 16; off > 0; off >>= 1)
      acc[e] += __shfl_xor(acc[e], off, 32);

  if (lane == 0) {
    float mx = acc[0];
#pragma unroll
    for (int e = 1; e < E_; ++e) mx = fmaxf(mx, acc[e]);
    float p[E_], s = 0.f;
#pragma unroll
    for (int e = 0; e < E_; ++e) { p[e] = __expf(acc[e] - mx); s += p[e]; }
    float inv = 1.f / s;
#pragma unroll
    for (int e = 0; e < E_; ++e) p[e] *= inv;
    int   i1 = 0;  float p1 = p[0];
#pragma unroll
    for (int e = 1; e < E_; ++e) if (p[e] > p1) { p1 = p[e]; i1 = e; }
    int   i2 = -1; float p2 = -1.f;
#pragma unroll
    for (int e = 0; e < E_; ++e) if (e != i1 && p[e] > p2) { p2 = p[e]; i2 = e; }
    size_t o = (size_t)(b * T_ + t) * TOPK;
    topk_p[o] = p1; topk_p[o + 1] = p2;
    topk_i[o] = i1; topk_i[o + 1] = i2;
  }
}

// ---------------------------------------------------------------------------
// 3) Deterministic slot assignment (k-major cumsum, matches reference order)
// ---------------------------------------------------------------------------
__global__ void moe_assign(const int* __restrict__ topk_i, int* __restrict__ pos) {
  int tid = threadIdx.x;
  if (tid >= B_ * E_) return;
  int b = tid >> 3, e = tid & 7;
  int cnt = 0;
  for (int k = 0; k < TOPK; ++k)
    for (int t = 0; t < T_; ++t) {
      int idx = (b * T_ + t) * TOPK + k;
      if (topk_i[idx] == e) pos[idx] = cnt++;
    }
}

// ---------------------------------------------------------------------------
// 4) Dispatch: scatter token rows (f32 -> bf16) into (E, B*CAP, C) buffer
// ---------------------------------------------------------------------------
__global__ __launch_bounds__(256) void moe_dispatch(
    const float* __restrict__ x, const int* __restrict__ topk_i,
    const int* __restrict__ pos, unsigned short* __restrict__ X16) {
  int a    = blockIdx.x * 8 + (threadIdx.x >> 5);
  int lane = threadIdx.x & 31;
  int b = a / (T_ * TOPK);
  int rem = a % (T_ * TOPK);
  int t = rem >> 1, k = rem & 1;
  int idx = (b * T_ + t) * TOPK + k;
  int e = topk_i[idx];
  int p = pos[idx];
  if (p >= CAP) return;                       // capacity overflow -> dropped
  const float* src = x + (size_t)(b * T_ + t) * C_;
  unsigned short* dst = X16 + (size_t)e * RPE * C_ + (size_t)(b * CAP + p) * C_;
#pragma unroll
  for (int i = 0; i < 8; ++i) {
    int c = lane * 4 + i * 128;
    float4 v = *(const float4*)(src + c);
    ushort4 o;
    o.x = f2bf(v.x); o.y = f2bf(v.y); o.z = f2bf(v.z); o.w = f2bf(v.w);
    *(ushort4*)(dst + c) = o;
  }
}

// ---------------------------------------------------------------------------
// shared tile staging helper (fallback path): 64x32 bf16 tile -> LDS stride 48
// ---------------------------------------------------------------------------
__device__ __forceinline__ void stage_tile(const unsigned short* __restrict__ g,
                                           size_t stride, int kt,
                                           unsigned short* __restrict__ s,
                                           int tid) {
#pragma unroll
  for (int i = 0; i < 2; ++i) {
    int chunk = tid + i * 128;
    int r = chunk >> 2, c8 = (chunk & 3) * 8;
    uint4 v = *(const uint4*)(g + (size_t)r * stride + (size_t)kt * 32 + c8);
    *(uint4*)(s + r * LDSTR + c8) = v;
  }
}

// ---------------------------------------------------------------------------
// 5) SwiGLU GEMM: P = silu(X*Wg) .* (X*Wfc).  bf16 WMMA, f32 accumulate.
//    Block 64x64, BK=32, 4 waves.  TDM double-buffered: wave0 issues tile kt+1
//    into the alternate LDS buffer, then s_wait_tensorcnt(3) fences only the
//    current tile's 3 loads (in-order per-wave completion) -> DMA overlaps WMMA.
// ---------------------------------------------------------------------------
__global__ __launch_bounds__(128) void moe_gemm_fc(
    const unsigned short* __restrict__ X16,
    const unsigned short* __restrict__ WfcT,   // [E][H][C] bf16 (n-major)
    const unsigned short* __restrict__ WgT,    // [E][H][C] bf16 (n-major)
    unsigned short* __restrict__ P16) {
  __shared__ __align__(16) unsigned short sA [2][TILE_E];
  __shared__ __align__(16) unsigned short sBf[2][TILE_E];
  __shared__ __align__(16) unsigned short sBg[2][TILE_E];
  constexpr int NK = C_ / 32;

  int e    = blockIdx.z;
  int row0 = blockIdx.y * 64;
  int col0 = blockIdx.x * 64;
  int tid  = threadIdx.x;
  int wv = tid >> 5, lane = tid & 31;
  int wm = (wv >> 1) * 32, wn = (wv & 1) * 32;
  int half = lane >> 4, lm = lane & 15;

  const unsigned short* Ab  = X16  + (size_t)e * RPE * C_ + (size_t)row0 * C_;
  const unsigned short* Bfb = WfcT + (size_t)e * H_ * C_  + (size_t)col0 * C_;
  const unsigned short* Bgb = WgT  + (size_t)e * H_ * C_  + (size_t)col0 * C_;

  v8f zero = {0.f,0.f,0.f,0.f,0.f,0.f,0.f,0.f};
  v8f accF[2][2], accG[2][2];
#pragma unroll
  for (int i = 0; i < 2; ++i)
#pragma unroll
    for (int j = 0; j < 2; ++j) { accF[i][j] = zero; accG[i][j] = zero; }

#ifdef USE_TDM
  // prologue: tile 0 -> buffer 0
  if (wv == 0) {
    tdm_load_tile((unsigned long long)(uintptr_t)(Ab),  (unsigned)(uintptr_t)(void*)sA[0],  C_);
    tdm_load_tile((unsigned long long)(uintptr_t)(Bfb), (unsigned)(uintptr_t)(void*)sBf[0], C_);
    tdm_load_tile((unsigned long long)(uintptr_t)(Bgb), (unsigned)(uintptr_t)(void*)sBg[0], C_);
  }
#else
  stage_tile(Ab,  C_, 0, sA[0],  tid);
  stage_tile(Bfb, C_, 0, sBf[0], tid);
  stage_tile(Bgb, C_, 0, sBg[0], tid);
#endif

  for (int kt = 0; kt < NK; ++kt) {
    int cur = kt & 1, nxt = (kt + 1) & 1;
#ifdef USE_TDM
    if (wv == 0) {
      if (kt + 1 < NK) {
        tdm_load_tile((unsigned long long)(uintptr_t)(Ab  + (size_t)(kt + 1) * 32),
                      (unsigned)(uintptr_t)(void*)sA[nxt],  C_);
        tdm_load_tile((unsigned long long)(uintptr_t)(Bfb + (size_t)(kt + 1) * 32),
                      (unsigned)(uintptr_t)(void*)sBf[nxt], C_);
        tdm_load_tile((unsigned long long)(uintptr_t)(Bgb + (size_t)(kt + 1) * 32),
                      (unsigned)(uintptr_t)(void*)sBg[nxt], C_);
        __builtin_amdgcn_s_wait_tensorcnt(3);   // current tile landed; next in flight
      } else {
        __builtin_amdgcn_s_wait_tensorcnt(0);   // last tile: drain
      }
    }
#else
    if (kt + 1 < NK) {
      stage_tile(Ab,  C_, kt + 1, sA[nxt],  tid);
      stage_tile(Bfb, C_, kt + 1, sBf[nxt], tid);
      stage_tile(Bgb, C_, kt + 1, sBg[nxt], tid);
    }
#endif
    __syncthreads();

    v16bf af[2], bf[2], bg[2];
#pragma unroll
    for (int mt = 0; mt < 2; ++mt)
      af[mt] = load_frag(sA[cur] + (wm + mt * 16 + lm) * LDSTR + 8 * half);
#pragma unroll
    for (int nt = 0; nt < 2; ++nt) {
      bf[nt] = load_frag(sBf[cur] + (wn + nt * 16 + lm) * LDSTR + 8 * half);
      bg[nt] = load_frag(sBg[cur] + (wn + nt * 16 + lm) * LDSTR + 8 * half);
    }
#pragma unroll
    for (int mt = 0; mt < 2; ++mt)
#pragma unroll
      for (int nt = 0; nt < 2; ++nt) {
        accF[mt][nt] = __builtin_amdgcn_wmma_f32_16x16x32_bf16(
            false, af[mt], false, bf[nt], (short)0, accF[mt][nt], false, false);
        accG[mt][nt] = __builtin_amdgcn_wmma_f32_16x16x32_bf16(
            false, af[mt], false, bg[nt], (short)0, accG[mt][nt], false, false);
      }
    __syncthreads();
  }

  // epilogue: silu(g) * h  -> bf16
  unsigned short* Pb = P16 + (size_t)e * RPE * H_;
#pragma unroll
  for (int mt = 0; mt < 2; ++mt)
#pragma unroll
    for (int nt = 0; nt < 2; ++nt) {
      int n = col0 + wn + nt * 16 + lm;
#pragma unroll
      for (int r = 0; r < 8; ++r) {
        int m = row0 + wm + mt * 16 + r + 8 * half;
        float h = accF[mt][nt][r];
        float g = accG[mt][nt][r];
        float sg = g / (1.f + __expf(-g));
        Pb[(size_t)m * H_ + n] = f2bf(sg * h);
      }
    }
}

// ---------------------------------------------------------------------------
// 6) Projection GEMM: O = P * WprojT  (K = H = 2048), f32 output.
//    Same double-buffered TDM pipeline, 2 loads/tile -> s_wait_tensorcnt(2).
// ---------------------------------------------------------------------------
__global__ __launch_bounds__(128) void moe_gemm_proj(
    const unsigned short* __restrict__ P16,
    const unsigned short* __restrict__ WpT,    // [E][C][H] bf16 (n-major)
    float* __restrict__ O32) {
  __shared__ __align__(16) unsigned short sA[2][TILE_E];
  __shared__ __align__(16) unsigned short sB[2][TILE_E];
  constexpr int NK = H_ / 32;

  int e    = blockIdx.z;
  int row0 = blockIdx.y * 64;
  int col0 = blockIdx.x * 64;
  int tid  = threadIdx.x;
  int wv = tid >> 5, lane = tid & 31;
  int wm = (wv >> 1) * 32, wn = (wv & 1) * 32;
  int half = lane >> 4, lm = lane & 15;

  const unsigned short* Ab = P16 + (size_t)e * RPE * H_ + (size_t)row0 * H_;
  const unsigned short* Bb = WpT + (size_t)e * C_ * H_  + (size_t)col0 * H_;

  v8f zero = {0.f,0.f,0.f,0.f,0.f,0.f,0.f,0.f};
  v8f acc[2][2];
#pragma unroll
  for (int i = 0; i < 2; ++i)
#pragma unroll
    for (int j = 0; j < 2; ++j) acc[i][j] = zero;

#ifdef USE_TDM
  if (wv == 0) {
    tdm_load_tile((unsigned long long)(uintptr_t)(Ab), (unsigned)(uintptr_t)(void*)sA[0], H_);
    tdm_load_tile((unsigned long long)(uintptr_t)(Bb), (unsigned)(uintptr_t)(void*)sB[0], H_);
  }
#else
  stage_tile(Ab, H_, 0, sA[0], tid);
  stage_tile(Bb, H_, 0, sB[0], tid);
#endif

  for (int kt = 0; kt < NK; ++kt) {
    int cur = kt & 1, nxt = (kt + 1) & 1;
#ifdef USE_TDM
    if (wv == 0) {
      if (kt + 1 < NK) {
        tdm_load_tile((unsigned long long)(uintptr_t)(Ab + (size_t)(kt + 1) * 32),
                      (unsigned)(uintptr_t)(void*)sA[nxt], H_);
        tdm_load_tile((unsigned long long)(uintptr_t)(Bb + (size_t)(kt + 1) * 32),
                      (unsigned)(uintptr_t)(void*)sB[nxt], H_);
        __builtin_amdgcn_s_wait_tensorcnt(2);   // current tile landed; next in flight
      } else {
        __builtin_amdgcn_s_wait_tensorcnt(0);   // last tile: drain
      }
    }
#else
    if (kt + 1 < NK) {
      stage_tile(Ab, H_, kt + 1, sA[nxt], tid);
      stage_tile(Bb, H_, kt + 1, sB[nxt], tid);
    }
#endif
    __syncthreads();

    v16bf af[2], bf[2];
#pragma unroll
    for (int mt = 0; mt < 2; ++mt)
      af[mt] = load_frag(sA[cur] + (wm + mt * 16 + lm) * LDSTR + 8 * half);
#pragma unroll
    for (int nt = 0; nt < 2; ++nt)
      bf[nt] = load_frag(sB[cur] + (wn + nt * 16 + lm) * LDSTR + 8 * half);
#pragma unroll
    for (int mt = 0; mt < 2; ++mt)
#pragma unroll
      for (int nt = 0; nt < 2; ++nt)
        acc[mt][nt] = __builtin_amdgcn_wmma_f32_16x16x32_bf16(
            false, af[mt], false, bf[nt], (short)0, acc[mt][nt], false, false);
    __syncthreads();
  }

  float* Ob = O32 + (size_t)e * RPE * C_;
#pragma unroll
  for (int mt = 0; mt < 2; ++mt)
#pragma unroll
    for (int nt = 0; nt < 2; ++nt) {
      int n = col0 + wn + nt * 16 + lm;
#pragma unroll
      for (int r = 0; r < 8; ++r) {
        int m = row0 + wm + mt * 16 + r + 8 * half;
        Ob[(size_t)m * C_ + n] = acc[mt][nt][r];
      }
    }
}

// ---------------------------------------------------------------------------
// 7) Weighted combine gather: y = sum_k p_k * O[e_k, b*CAP+pos_k, :]
// ---------------------------------------------------------------------------
__global__ __launch_bounds__(256) void moe_combine(
    const float* __restrict__ O32, const float* __restrict__ topk_p,
    const int* __restrict__ topk_i, const int* __restrict__ pos,
    float* __restrict__ y) {
  int wave = blockIdx.x * 8 + (threadIdx.x >> 5);
  int lane = threadIdx.x & 31;
  int b = wave / T_, t = wave % T_;
  size_t o = (size_t)(b * T_ + t) * TOPK;
  float w0 = topk_p[o], w1 = topk_p[o + 1];
  int e0 = topk_i[o], e1 = topk_i[o + 1];
  int p0 = pos[o],    p1 = pos[o + 1];
  const float* r0 = (p0 < CAP) ? O32 + (size_t)e0 * RPE * C_ + (size_t)(b * CAP + p0) * C_ : nullptr;
  const float* r1 = (p1 < CAP) ? O32 + (size_t)e1 * RPE * C_ + (size_t)(b * CAP + p1) * C_ : nullptr;
  float* yo = y + (size_t)(b * T_ + t) * C_;
#pragma unroll
  for (int i = 0; i < 8; ++i) {
    int c = lane * 4 + i * 128;
    float4 a = {0.f, 0.f, 0.f, 0.f};
    if (r0) {
      float4 v = *(const float4*)(r0 + c);
      a.x = fmaf(w0, v.x, a.x); a.y = fmaf(w0, v.y, a.y);
      a.z = fmaf(w0, v.z, a.z); a.w = fmaf(w0, v.w, a.w);
    }
    if (r1) {
      float4 v = *(const float4*)(r1 + c);
      a.x = fmaf(w1, v.x, a.x); a.y = fmaf(w1, v.y, a.y);
      a.z = fmaf(w1, v.z, a.z); a.w = fmaf(w1, v.w, a.w);
    }
    *(float4*)(yo + c) = a;
  }
}

// ---------------------------------------------------------------------------
extern "C" void kernel_launch(void* const* d_in, const int* in_sizes, int n_in,
                              void* d_out, int out_size, void* d_ws, size_t ws_size,
                              hipStream_t stream) {
  const float* x   = (const float*)d_in[0];
  const float* wr  = (const float*)d_in[1];
  const float* wfc = (const float*)d_in[2];
  const float* wg  = (const float*)d_in[3];
  const float* wp  = (const float*)d_in[4];
  float* y = (float*)d_out;

  char* ws = (char*)d_ws;
  size_t off = 0;
  auto alloc = [&](size_t bytes) {
    char* p = ws + off;
    off = (off + bytes + 255) & ~(size_t)255;
    return p;
  };
  float*          topk_p = (float*)alloc(sizeof(float) * B_ * T_ * TOPK);
  int*            topk_i = (int*)  alloc(sizeof(int)   * B_ * T_ * TOPK);
  int*            pos    = (int*)  alloc(sizeof(int)   * B_ * T_ * TOPK);
  unsigned short* WfcT   = (unsigned short*)alloc((size_t)2 * E_ * C_ * H_);
  unsigned short* WgT    = (unsigned short*)alloc((size_t)2 * E_ * C_ * H_);
  unsigned short* WpT    = (unsigned short*)alloc((size_t)2 * E_ * H_ * C_);
  unsigned short* X16    = (unsigned short*)alloc((size_t)2 * E_ * RPE * C_);
  unsigned short* P16    = (unsigned short*)alloc((size_t)2 * E_ * RPE * H_);
  float*          O32    = (float*)         alloc((size_t)4 * E_ * RPE * C_);
  (void)ws_size; (void)in_sizes; (void)n_in; (void)out_size;

  // weight conversion + transpose to [n][k] bf16 (one-time per call)
  moe_cvt_bf16_T<<<dim3(H_ / 32, C_ / 32, E_), 256, 0, stream>>>(wfc, WfcT, C_, H_);
  moe_cvt_bf16_T<<<dim3(H_ / 32, C_ / 32, E_), 256, 0, stream>>>(wg,  WgT,  C_, H_);
  moe_cvt_bf16_T<<<dim3(C_ / 32, H_ / 32, E_), 256, 0, stream>>>(wp,  WpT,  H_, C_);

  moe_router  <<<B_ * T_ / 8, 256, 0, stream>>>(x, wr, topk_p, topk_i);
  moe_assign  <<<1, 64, 0, stream>>>(topk_i, pos);
  moe_dispatch<<<B_ * T_ * TOPK / 8, 256, 0, stream>>>(x, topk_i, pos, X16);

  moe_gemm_fc  <<<dim3(H_ / 64, RPE / 64, E_), 128, 0, stream>>>(X16, WfcT, WgT, P16);
  moe_gemm_proj<<<dim3(C_ / 64, RPE / 64, E_), 128, 0, stream>>>(P16, WpT, O32);

  moe_combine<<<B_ * T_ / 8, 256, 0, stream>>>(O32, topk_p, topk_i, pos, y);
}
